// HyperbolicRotH_69303592288542
// MI455X (gfx1250) — compile-verified, matrix operand order checked by reading, not run on hardware
//
#include <hip/hip_runtime.h>
#include <math.h>

// ---------------- problem constants ---------------------------------------
#define DIM      512
#define E_TOT    40960
#define BQ       128          // number of queries (triplets)
#define CC       0.01f        // curvature c
#define SQRT_C   0.1f
#define EPSF     1e-15f
#define TCLIP    (1.0f - 1e-6f)

#define NTILE_BLK 64          // entity rows per block (4 n-tiles of 16)
#define QA_DWORDS 32768       // 8 mtiles * 16 ksteps * 32 lanes * 8 dwords (per precision)
#define FSTRIDE   12          // per-lane dword stride of LDS B-fragments (bank-conflict-free)
#define BFRAG_DW  (4 * 16 * 32 * FSTRIDE)   // 24576 dwords per precision

// ---------------- async-to-LDS (signature learned from probe diagnostic) ---
#define TRY_ASYNC 1
#if TRY_ASYNC && __has_builtin(__builtin_amdgcn_global_load_async_to_lds_b128)
#define HAS_ASYNC 1
typedef __attribute__((ext_vector_type(4))) int v4i;
typedef __attribute__((address_space(1))) v4i* as1v4i;   // global-side arg
typedef __attribute__((address_space(3))) v4i* as3v4i;   // LDS-side arg
#endif

typedef __attribute__((ext_vector_type(16))) __bf16 v16bf;
typedef __attribute__((ext_vector_type(8)))  float  v8f;

union FragB16 { unsigned int u[8]; v16bf v; };

// f32 -> bf16 (RNE) used only in the tiny query kernel
__device__ __forceinline__ unsigned short f2bf(float f) {
    unsigned int u = __float_as_uint(f);
    u += 0x7FFFu + ((u >> 16) & 1u);
    return (unsigned short)(u >> 16);
}
__device__ __forceinline__ float bf2f(unsigned short b) {
    return __uint_as_float(((unsigned int)b) << 16);
}
__device__ __forceinline__ unsigned int pack2(unsigned short lo, unsigned short hi) {
    return (unsigned int)lo | ((unsigned int)hi << 16);
}

// -------------------------------------------------------------------------
// Kernel 1: queries (log-map -> Givens -> exp-map -> mobius_add), written in
// WMMA A-fragment layout as bf16 hi/lo split. One block per triplet.
// -------------------------------------------------------------------------
__device__ __forceinline__ float block_reduce(float v, float* red, int t) {
    __syncthreads();
    red[t] = v;
    __syncthreads();
    for (int s = 128; s > 0; s >>= 1) {
        if (t < s) red[t] += red[t + s];
        __syncthreads();
    }
    return red[0];
}

__global__ __launch_bounds__(256) void rot_h_queries(
    const float* __restrict__ ent,
    const float* __restrict__ rel_rot,
    const float* __restrict__ rel_trans,
    const float* __restrict__ ebias,
    const int*   __restrict__ trip,
    unsigned int* __restrict__ qA,
    float* __restrict__ q2_out,
    float* __restrict__ sbias_out)
{
    __shared__ float red[256];
    const int b = blockIdx.x;
    const int t = threadIdx.x;
    const int s_idx = trip[3 * b + 0];
    const int r_idx = trip[3 * b + 1];

    float x1 = ent[(size_t)s_idx * DIM + 2 * t];
    float x2 = ent[(size_t)s_idx * DIM + 2 * t + 1];

    // log_map_zero
    float ss = block_reduce(x1 * x1 + x2 * x2, red, t);
    float sn = fmaxf(sqrtf(ss), EPSF);
    float lfac = atanhf(fminf(SQRT_C * sn, TCLIP)) / (SQRT_C * sn);
    x1 *= lfac; x2 *= lfac;

    // Givens rotation (thread owns one even/odd pair)
    float ang = rel_rot[(size_t)r_idx * (DIM / 2) + t];
    float sa, ca;
    sincosf(ang, &sa, &ca);
    float o1 = ca * x1 - sa * x2;
    float o2 = sa * x1 + ca * x2;

    // exp_map_zero(rot)
    float vs = block_reduce(o1 * o1 + o2 * o2, red, t);
    float vn = fmaxf(sqrtf(vs), EPSF);
    float ef = tanhf(SQRT_C * vn) / (SQRT_C * vn);
    float rx = o1 * ef, ry = o2 * ef;

    // t_r = exp_map_zero(rel_trans[r])
    float t1 = rel_trans[(size_t)r_idx * DIM + 2 * t];
    float t2 = rel_trans[(size_t)r_idx * DIM + 2 * t + 1];
    float ts = block_reduce(t1 * t1 + t2 * t2, red, t);
    float tn = fmaxf(sqrtf(ts), EPSF);
    float tf = tanhf(SQRT_C * tn) / (SQRT_C * tn);
    t1 *= tf; t2 *= tf;

    // mobius_add(rot_s, t_r)
    float x2s = block_reduce(rx * rx + ry * ry, red, t);
    float y2s = block_reduce(t1 * t1 + t2 * t2, red, t);
    float xys = block_reduce(rx * t1 + ry * t2, red, t);
    float Af  = 1.0f + 2.0f * CC * xys + CC * y2s;
    float Bf  = 1.0f - CC * x2s;
    float den = fmaxf(1.0f + 2.0f * CC * xys + CC * CC * x2s * y2s, EPSF);
    float q1 = (Af * rx + Bf * t1) / den;
    float q2 = (Af * ry + Bf * t2) / den;

    float qq = block_reduce(q1 * q1 + q2 * q2, red, t);
    if (t == 0) { q2_out[b] = qq; sbias_out[b] = ebias[s_idx]; }

    // A-fragment layout: lane L holds M=L%16, K-half=L/16;
    // dword j holds pair k = (j<4?0:16) + (L/16)*8 + (j&3)*2
    const int K   = 2 * t;
    const int kst = K >> 5;
    const int kin = K & 31;
    const int hf  = (kin & 15) >> 3;
    const int jj  = ((kin & 16) >> 2) + ((kin & 7) >> 1);
    const int lane  = hf * 16 + (b & 15);
    const int mtile = b >> 4;
    const int idx = (((mtile * 16 + kst) * 32) + lane) * 8 + jj;

    unsigned short h1 = f2bf(q1), h2 = f2bf(q2);
    qA[idx]             = pack2(h1, h2);
    qA[QA_DWORDS + idx] = pack2(f2bf(q1 - bf2f(h1)), f2bf(q2 - bf2f(h2)));
}

// -------------------------------------------------------------------------
// Kernel 2: split-precision bf16 WMMA GEMM + hyperbolic epilogue.
// Staging pass converts the 64-row entity tile to bf16 hi/lo ONCE and stores
// it in LDS directly in WMMA B-fragment layout (stride-12 => conflict-free
// ds_load_b128). K-loop is then pure LDS loads + WMMA.
// -------------------------------------------------------------------------
__global__ __launch_bounds__(256) void rot_h_scores(
    const float* __restrict__ ent,
    const float* __restrict__ ebias,
    const unsigned int* __restrict__ qA,
    const float* __restrict__ q2v,
    const float* __restrict__ sbias,
    float* __restrict__ out)
{
    extern __shared__ unsigned int smu[];
    unsigned int* BHI = smu;                    // [BFRAG_DW]
    unsigned int* BLO = smu + BFRAG_DW;         // [BFRAG_DW]
    float* e2p = (float*)(smu + 2 * BFRAG_DW);  // [256]
    float* e2s = e2p + 256;                     // [64]
    float* ebs = e2s + 64;                      // [64] (16B-aligned offset)

    const int tid = threadIdx.x;
    const int n0  = blockIdx.x * NTILE_BLK;

#if defined(HAS_ASYNC)
    // CDNA5 async copy: stage the 64 entity biases for this block into LDS.
    if (tid < 16) {
        as1v4i g = (as1v4i)(unsigned long long)(ebias + n0 + 4 * tid);
        as3v4i l = (as3v4i)(unsigned int)(unsigned long long)(ebs + 4 * tid);
        __builtin_amdgcn_global_load_async_to_lds_b128(g, l, 0, 0);
    }
#endif

    // ---- staging: fp32 rows -> bf16 hi/lo B-fragments in LDS ------------
    // thread t: row = t/4, quarter-row = t%4 (contiguous 512B stream/lane)
    {
        const int srow  = tid >> 2;
        const int sq    = tid & 3;
        const float* grow = ent + (size_t)(n0 + srow) * DIM + sq * 128;
        const int ntS   = srow >> 4;
        const int lrowS = srow & 15;
        float e2acc = 0.0f;
        #pragma unroll 4
        for (int it = 0; it < 32; ++it) {
            const float4 g = *(const float4*)(grow + it * 4);
            e2acc += g.x * g.x + g.y * g.y + g.z * g.z + g.w * g.w;

            const int k    = sq * 128 + it * 4;
            const int ks   = k >> 5;
            const int kin  = k & 31;
            const int half = (kin & 15) >> 3;
            const int j0   = ((kin & 16) >> 2) + ((kin & 7) >> 1);  // in {0,2,4,6}
            const int lane = half * 16 + lrowS;
            const int base = ((ntS * 16 + ks) * 32 + lane) * FSTRIDE + j0;

            const unsigned ux = __float_as_uint(g.x), uy = __float_as_uint(g.y);
            const unsigned uz = __float_as_uint(g.z), uw = __float_as_uint(g.w);
            // hi = truncate-to-bf16; pack two -> one dword with one v_perm_b32
            const unsigned h0 = __builtin_amdgcn_perm(uy, ux, 0x07060302u);
            const unsigned h1 = __builtin_amdgcn_perm(uw, uz, 0x07060302u);
            // lo = residual (x - hi), truncated to bf16
            const float lx = g.x - __uint_as_float(ux & 0xffff0000u);
            const float ly = g.y - __uint_as_float(uy & 0xffff0000u);
            const float lz = g.z - __uint_as_float(uz & 0xffff0000u);
            const float lw = g.w - __uint_as_float(uw & 0xffff0000u);
            const unsigned l0 = __builtin_amdgcn_perm(__float_as_uint(ly), __float_as_uint(lx), 0x07060302u);
            const unsigned l1 = __builtin_amdgcn_perm(__float_as_uint(lw), __float_as_uint(lz), 0x07060302u);

            *(uint2*)&BHI[base] = make_uint2(h0, h1);
            *(uint2*)&BLO[base] = make_uint2(l0, l1);
        }
        e2p[tid] = e2acc;   // fused y2 = ||e||^2 partial (saves an HBM pass)
    }
    __syncthreads();
    if (tid < NTILE_BLK)
        e2s[tid] = e2p[4 * tid] + e2p[4 * tid + 1] + e2p[4 * tid + 2] + e2p[4 * tid + 3];
#if defined(HAS_ASYNC)
#if __has_builtin(__builtin_amdgcn_s_wait_asynccnt)
    __builtin_amdgcn_s_wait_asynccnt(0);
#else
    asm volatile("s_wait_asynccnt 0" ::: "memory");
#endif
#else
    if (tid < NTILE_BLK) ebs[tid] = ebias[n0 + tid];
#endif
    __syncthreads();

    // ---- K loop: pure ds_load_b128 + WMMA -------------------------------
    const int lane  = tid & 31;
    const int mtile = tid >> 5;     // wave id == M-tile id
    const int hf    = lane >> 4;
    const int nl    = lane & 15;

    v8f acc[4];
    for (int nt = 0; nt < 4; ++nt)
        for (int i = 0; i < 8; ++i) acc[nt][i] = 0.0f;

    for (int ks = 0; ks < 16; ++ks) {
        FragB16 ahi, alo;
        const unsigned int* abase = qA + (((mtile * 16 + ks) * 32) + lane) * 8;
        if (ks + 1 < 16)
            __builtin_prefetch(qA + (((mtile * 16 + ks + 1) * 32) + lane) * 8, 0, 0);
        *(uint4*)&ahi.u[0] = *(const uint4*)(abase);
        *(uint4*)&ahi.u[4] = *(const uint4*)(abase + 4);
        *(uint4*)&alo.u[0] = *(const uint4*)(abase + QA_DWORDS);
        *(uint4*)&alo.u[4] = *(const uint4*)(abase + QA_DWORDS + 4);

        #pragma unroll
        for (int nt = 0; nt < 4; ++nt) {
            const int fb = ((nt * 16 + ks) * 32 + lane) * FSTRIDE;
            FragB16 bh, bl;
            *(uint4*)&bh.u[0] = *(const uint4*)&BHI[fb];
            *(uint4*)&bh.u[4] = *(const uint4*)&BHI[fb + 4];
            *(uint4*)&bl.u[0] = *(const uint4*)&BLO[fb];
            *(uint4*)&bl.u[4] = *(const uint4*)&BLO[fb + 4];

            acc[nt] = __builtin_amdgcn_wmma_f32_16x16x32_bf16(
                false, ahi.v, false, bh.v, (short)0, acc[nt], false, false);
            acc[nt] = __builtin_amdgcn_wmma_f32_16x16x32_bf16(
                false, ahi.v, false, bl.v, (short)0, acc[nt], false, false);
            acc[nt] = __builtin_amdgcn_wmma_f32_16x16x32_bf16(
                false, alo.v, false, bh.v, (short)0, acc[nt], false, false);
        }
    }

    // ---- hyperbolic epilogue -------------------------------------------
    float q2l[8], sbl[8];
    #pragma unroll
    for (int i = 0; i < 8; ++i) {
        const int m = mtile * 16 + hf * 8 + i;
        q2l[i] = q2v[m];
        sbl[i] = sbias[m];
    }
    for (int nt = 0; nt < 4; ++nt) {
        const int n  = nt * 16 + nl;
        const float y2 = e2s[n];
        const float eb = ebs[n];
        #pragma unroll
        for (int i = 0; i < 8; ++i) {
            const int m    = mtile * 16 + hf * 8 + i;
            const float xy = -acc[nt][i];          // x = -query
            const float x2 = q2l[i];
            const float Af  = 1.0f + 2.0f * CC * xy + CC * y2;
            const float Bf  = 1.0f - CC * x2;
            const float den = fmaxf(1.0f + 2.0f * CC * xy + CC * CC * x2 * y2, EPSF);
            const float num2 = Af * Af * x2 + 2.0f * Af * Bf * xy + Bf * Bf * y2;
            out[(size_t)m * E_TOT + (n0 + n)] = -(num2 / (den * den)) + eb + sbl[i];
        }
    }
}

// -------------------------------------------------------------------------
extern "C" void kernel_launch(void* const* d_in, const int* in_sizes, int n_in,
                              void* d_out, int out_size, void* d_ws, size_t ws_size,
                              hipStream_t stream) {
    const float* ent    = (const float*)d_in[0];
    // d_in[1] = rel_embedding (unused by reference)
    const float* rrot   = (const float*)d_in[2];
    const float* rtrans = (const float*)d_in[3];
    const float* ebias  = (const float*)d_in[4];
    const int*   trip   = (const int*)d_in[5];
    float* out = (float*)d_out;

    unsigned int* qA = (unsigned int*)d_ws;
    float* q2v = (float*)((char*)d_ws + 2 * QA_DWORDS * sizeof(unsigned int));
    float* sb  = q2v + BQ;

    rot_h_queries<<<BQ, 256, 0, stream>>>(ent, rrot, rtrans, ebias, trip, qA, q2v, sb);

    const size_t shmem = (2 * BFRAG_DW + 256 + 64 + 64) * sizeof(unsigned int);
    (void)hipFuncSetAttribute((const void*)rot_h_scores,
                              hipFuncAttributeMaxDynamicSharedMemorySize, (int)shmem);
    rot_h_scores<<<E_TOT / NTILE_BLK, 256, shmem, stream>>>(ent, ebias, qA, q2v, sb, out);
}
